// GPT_4088808866411
// MI455X (gfx1250) — matrix operand order repaired and checked by reading, not compiled, not run on hardware
//
#include <hip/hip_runtime.h>
#include <hip/hip_bf16.h>
#include <math.h>

typedef __attribute__((ext_vector_type(16))) _Float16 v16h;
typedef __attribute__((ext_vector_type(8)))  _Float16 v8h;
typedef __attribute__((ext_vector_type(8)))  float    v8f;
typedef __attribute__((ext_vector_type(4)))  int      v4i;

#define NB   4
#define TT   1024
#define DD   768
#define LLY  8
#define HH   12
#define DHH  64
#define FFN  3072
#define VV   32000
#define NTOK (NB*TT)          // 4096
#define ATT_SCALE 0.125f      // 1/sqrt(64)
#define LN_EPS 1e-5f

// GEMM tiling: block = 256 threads = 8 waves; block tile 128(M) x 64(N);
// each wave owns a 16x64 strip; B tile (64 x 32 halves) staged in LDS.
#define MBLK 128
#define NBLK 64
#define KSTEP 32
#define BROW 40               // padded LDS row stride in halves (80B)
#define BBUF (NBLK * BROW)    // halves per LDS buffer

#if defined(__has_builtin)
#  if __has_builtin(__builtin_amdgcn_global_load_async_to_lds_b128) && \
      __has_builtin(__builtin_amdgcn_s_wait_asynccnt)
#    define GPT_ASYNC_LDS 1
#  endif
#endif
#ifndef GPT_ASYNC_LDS
#  define GPT_ASYNC_LDS 0
#endif

// ---------------------------------------------------------------- reductions
__device__ __forceinline__ float block_reduce_sum256(float v, float* red) {
    int tid = threadIdx.x;
    red[tid] = v; __syncthreads();
    for (int s = 128; s > 0; s >>= 1) {
        if (tid < s) red[tid] += red[tid + s];
        __syncthreads();
    }
    float r = red[0]; __syncthreads();
    return r;
}
__device__ __forceinline__ float block_reduce_max256(float v, float* red) {
    int tid = threadIdx.x;
    red[tid] = v; __syncthreads();
    for (int s = 128; s > 0; s >>= 1) {
        if (tid < s) red[tid] = fmaxf(red[tid], red[tid + s]);
        __syncthreads();
    }
    float r = red[0]; __syncthreads();
    return r;
}

// ---------------------------------------------------------------- embedding
__global__ void embed_kernel(const int* __restrict__ idx,
                             const float* __restrict__ tok_emb,
                             const float* __restrict__ pos_emb,
                             float* __restrict__ x) {
    long i = (long)blockIdx.x * blockDim.x + threadIdx.x;
    if (i >= (long)NTOK * DD) return;
    int d   = (int)(i % DD);
    int row = (int)(i / DD);          // b*T + t
    int t   = row % TT;
    int tok = idx[row];
    x[i] = tok_emb[(long)tok * DD + d] + pos_emb[(long)t * DD + d];
}

// ---------------------------------------------------------------- f32 -> f16
__global__ void cvt_f32_f16(const float* __restrict__ src,
                            _Float16* __restrict__ dst, long n) {
    long i = (long)blockIdx.x * blockDim.x + threadIdx.x;
    if (i < n) dst[i] = (_Float16)src[i];
}

// ---------------------------------------------------------------- layernorm (row = 768), writes f16
__global__ void ln_kernel(const float* __restrict__ x,
                          const float* __restrict__ w,
                          const float* __restrict__ b,
                          _Float16* __restrict__ out) {
    __shared__ float red[256];
    int row = blockIdx.x;
    int tid = threadIdx.x;
    const float* xr = x + (long)row * DD;
    float s = 0.f;
    for (int i = tid; i < DD; i += 256) s += xr[i];
    float mu = block_reduce_sum256(s, red) * (1.0f / DD);
    float v = 0.f;
    for (int i = tid; i < DD; i += 256) { float d = xr[i] - mu; v += d * d; }
    float var  = block_reduce_sum256(v, red) * (1.0f / DD);
    float rstd = rsqrtf(var + LN_EPS);
    _Float16* orow = out + (long)row * DD;
    for (int i = tid; i < DD; i += 256)
        orow[i] = (_Float16)((xr[i] - mu) * rstd * w[i] + b[i]);
}

// ---------------------------------------------------------------- WMMA GEMM
// C[M,N] = A[M,K](f16) * W[N,K](f16)^T (+bias) (gelu?) (+residual)
// Block tile 128x64: B tile shared through LDS (async double-buffered via
// GLOBAL_LOAD_ASYNC_TO_LDS_B128), A streamed per wave from global.
#if GPT_ASYNC_LDS
typedef __attribute__((address_space(3))) _Float16 lds_h;
__device__ __forceinline__ void async_cp16(const _Float16* g, lds_h* l) {
    __builtin_amdgcn_global_load_async_to_lds_b128(
        (__attribute__((address_space(1))) v4i*)g,
        (__attribute__((address_space(3))) v4i*)l, 0, 0);
}
#endif

__global__ __launch_bounds__(256)
void wmma_gemm(const _Float16* __restrict__ A,
               const _Float16* __restrict__ W,
               const float* __restrict__ bias,       // [N] or null
               const float* __restrict__ residual,   // [M,ldc] or null
               float* __restrict__ outF,             // [M,ldc] or null
               _Float16* __restrict__ outH,          // [M,ldh] or null
               int M, int N, int K, int ldc, int ldh, int gelu) {
    __shared__ _Float16 Bs[2 * BBUF];

    int tid  = threadIdx.x;
    int wave = tid >> 5, lane = tid & 31;
    int tilesN = N / NBLK;
    int bm = blockIdx.x / tilesN, bn = blockIdx.x % tilesN;
    int m0 = bm * MBLK + wave * 16;
    int n0 = bn * NBLK;
    int r  = lane & 15, hs = lane >> 4;

    // LDS staging: each of 256 threads moves 16B; row = tid/4, col = (tid%4)*8
    int srow = tid >> 2, scol = (tid & 3) << 3;
    const _Float16* gsrc = W + (long)(n0 + srow) * K + scol;   // + k at use

    const _Float16* arow = A + (long)(m0 + r) * K;
    const int nk = K / KSTEP;
    v8f cacc[4] = {};

#if GPT_ASYNC_LDS
    // AS3-typed base hoisted out of the loop: one addrspacecast, offset math in AS3
    lds_h* lbase = (lds_h*)&Bs[srow * BROW + scol];
    // prime two k-steps
    async_cp16(gsrc, lbase);
    if (nk > 1) async_cp16(gsrc + KSTEP, lbase + BBUF);
    for (int ki = 0; ki < nk; ++ki) {
        int cur = ki & 1;
        if (ki + 1 < nk) __builtin_amdgcn_s_wait_asynccnt(1);
        else             __builtin_amdgcn_s_wait_asynccnt(0);
        __syncthreads();                       // B tile 'cur' visible to all waves
#else
    for (int ki = 0; ki < nk; ++ki) {
        int cur = 0;
        v8h breg = *(const v8h*)(gsrc + ki * KSTEP);
        __syncthreads();                       // previous step done reading Bs
        *(v8h*)&Bs[srow * BROW + scol] = breg;
        __syncthreads();                       // B tile staged
#endif
        int k0 = ki * KSTEP;
        // A fragment per ISA layout: lane<16 -> K {0..7,16..23}; lane>=16 -> {8..15,24..31}
        v8h lo = *(const v8h*)(arow + k0 + 8 * hs);
        v8h hi = *(const v8h*)(arow + k0 + 16 + 8 * hs);
        if (ki + 1 < nk) __builtin_prefetch(arow + k0 + KSTEP, 0, 3);
        v16h afrag;
#pragma unroll
        for (int i = 0; i < 8; ++i) { afrag[i] = lo[i]; afrag[8 + i] = hi[i]; }

        const _Float16* bbase = &Bs[cur * BBUF + r * BROW + 8 * hs];
#pragma unroll
        for (int s = 0; s < 4; ++s) {
            v8h b0 = *(const v8h*)(bbase + s * 16 * BROW);
            v8h b1 = *(const v8h*)(bbase + s * 16 * BROW + 16);
            v16h bfrag;
#pragma unroll
            for (int i = 0; i < 8; ++i) { bfrag[i] = b0[i]; bfrag[8 + i] = b1[i]; }
            cacc[s] = __builtin_amdgcn_wmma_f32_16x16x32_f16(
                false, afrag, false, bfrag, (short)0, cacc[s], false, false);
        }
#if GPT_ASYNC_LDS
        __syncthreads();                       // all waves done reading 'cur'
        if (ki + 2 < nk) async_cp16(gsrc + (ki + 2) * KSTEP, lbase + cur * BBUF);
#endif
    }

    // epilogue: C layout lane l -> N = 16*s + (l&15); VGPR v -> M = v + 8*(l>>4)
#pragma unroll
    for (int s = 0; s < 4; ++s) {
        int n = n0 + (s << 4) + r;
        float bv = bias ? bias[n] : 0.f;
#pragma unroll
        for (int v = 0; v < 8; ++v) {
            int m = m0 + v + 8 * hs;
            float val = cacc[s][v] + bv;
            if (gelu) val = 0.5f * val * (1.0f + erff(val * 0.70710678118654752f));
            if (residual) val += residual[(long)m * ldc + n];
            if (outF) outF[(long)m * ldc + n] = val;
            if (outH) outH[(long)m * ldh + n] = (_Float16)val;
        }
    }
}

// ---------------------------------------------------------------- attention
// grid = (T, H, B); block = 256. Streaming softmax(QK^T)*V for one (b,h,q) row.
__global__ __launch_bounds__(256)
void attn_kernel(const float* __restrict__ qkv,   // [NTOK, 3*D]
                 _Float16* __restrict__ outH) {   // [NTOK, D] f16
    __shared__ float qv[DHH];
    __shared__ float sc[TT];
    __shared__ float red[256];
    __shared__ float part[4][DHH];
    __shared__ float s_max, s_sum;

    int q = blockIdx.x, h = blockIdx.y, b = blockIdx.z;
    int tid = threadIdx.x;
    const long stride = 3L * DD;
    const long base   = ((long)b * TT) * stride;

    if (tid < DHH) qv[tid] = qkv[base + (long)q * stride + h * DHH + tid];
    __syncthreads();

    for (int t = tid; t <= q; t += 256) {
        const float* kr = qkv + base + (long)t * stride + DD + h * DHH;
        float s = 0.f;
#pragma unroll 8
        for (int d = 0; d < DHH; ++d) s += qv[d] * kr[d];
        sc[t] = s * ATT_SCALE;
    }
    __syncthreads();

    float m = -1e30f;
    for (int t = tid; t <= q; t += 256) m = fmaxf(m, sc[t]);
    m = block_reduce_max256(m, red);
    if (tid == 0) s_max = m;
    __syncthreads();

    float ssum = 0.f;
    for (int t = tid; t <= q; t += 256) {
        float e = __expf(sc[t] - s_max);
        sc[t] = e; ssum += e;
    }
    ssum = block_reduce_sum256(ssum, red);
    if (tid == 0) s_sum = ssum;
    __syncthreads();

    int g = tid >> 6, d = tid & 63;
    float acc = 0.f;
    for (int t = g; t <= q; t += 4)
        acc += sc[t] * qkv[base + (long)t * stride + 2 * DD + h * DHH + d];
    part[g][d] = acc;
    __syncthreads();
    if (g == 0) {
        float y = (part[0][d] + part[1][d] + part[2][d] + part[3][d]) / s_sum;
        outH[((long)b * TT + q) * DD + h * DHH + d] = (_Float16)y;
    }
}

// ---------------------------------------------------------------- loss
__global__ void loss_init(float* acc) { acc[0] = 0.f; acc[1] = 0.f; }

__global__ void loss_rows(const float* __restrict__ logits,   // [NTOK, V]
                          const int* __restrict__ targets,
                          float* __restrict__ acc) {
    __shared__ float red[256];
    int row = blockIdx.x, tid = threadIdx.x;
    const float* lr = logits + (long)row * VV;
    float m = -1e30f;
    for (int i = tid; i < VV; i += 256) m = fmaxf(m, lr[i]);
    m = block_reduce_max256(m, red);
    float s = 0.f;
    for (int i = tid; i < VV; i += 256) s += __expf(lr[i] - m);
    s = block_reduce_sum256(s, red);
    if (tid == 0) {
        int tgt = targets[row];
        if (tgt != -100) {
            float nll = -(lr[tgt] - m - logf(s));
            atomicAdd(&acc[0], nll);
            atomicAdd(&acc[1], 1.0f);
        }
    }
}

__global__ void loss_final(const float* __restrict__ acc, float* __restrict__ out) {
    out[0] = acc[0] / fmaxf(acc[1], 1.0f);
}

// ---------------------------------------------------------------- host
static inline int gemm_blocks(int M, int N) {
    return (M / MBLK) * (N / NBLK);
}

extern "C" void kernel_launch(void* const* d_in, const int* in_sizes, int n_in,
                              void* d_out, int out_size, void* d_ws, size_t ws_size,
                              hipStream_t stream) {
    const int*   idx      = (const int*)  d_in[0];
    const int*   targets  = (const int*)  d_in[1];
    const float* tok_emb  = (const float*)d_in[2];
    const float* pos_emb  = (const float*)d_in[3];
    const float* qkv_w    = (const float*)d_in[4];
    const float* qkv_b    = (const float*)d_in[5];
    const float* proj_w   = (const float*)d_in[6];
    const float* proj_b   = (const float*)d_in[7];
    const float* fc1_w    = (const float*)d_in[8];
    const float* fc1_b    = (const float*)d_in[9];
    const float* fc2_w    = (const float*)d_in[10];
    const float* fc2_b    = (const float*)d_in[11];
    const float* ln1_w    = (const float*)d_in[12];
    const float* ln1_b    = (const float*)d_in[13];
    const float* ln2_w    = (const float*)d_in[14];
    const float* ln2_b    = (const float*)d_in[15];
    const float* lnf_w    = (const float*)d_in[16];
    const float* lnf_b    = (const float*)d_in[17];

    float* logits = (float*)d_out;                       // [NTOK, V]
    float* lossO  = (float*)d_out + (long)NTOK * VV;     // scalar

    // workspace carve-up (256B aligned)
    char* ws = (char*)d_ws;
    size_t off = 0;
    auto carve = [&](size_t bytes) -> char* {
        char* p = ws + off;
        off = (off + bytes + 255) & ~(size_t)255;
        return p;
    };
    float*    X    = (float*)   carve((size_t)NTOK * DD * 4);        // residual stream
    _Float16* HA   = (_Float16*)carve((size_t)NTOK * DD * 2);        // ln/attn f16 activations
    _Float16* HB   = (_Float16*)carve((size_t)NTOK * FFN * 2);       // gelu f16 activations
    float*    QKV  = (float*)   carve((size_t)NTOK * 3 * DD * 4);    // qkv f32
    _Float16* WL   = (_Float16*)carve((size_t)FFN * DD * 2);         // per-layer weight f16
    _Float16* WEMB = (_Float16*)carve((size_t)VV * DD * 2);          // lm head f16
    float*    ACC  = (float*)   carve(256);
    if (off > ws_size) return;  // scratch too small; nothing safe to do

    const int TPB = 256;

    // embedding
    {
        long n = (long)NTOK * DD;
        embed_kernel<<<(int)((n + TPB - 1) / TPB), TPB, 0, stream>>>(idx, tok_emb, pos_emb, X);
    }

    for (int l = 0; l < LLY; ++l) {
        // --- attention half ---
        ln_kernel<<<NTOK, TPB, 0, stream>>>(X, ln1_w + (long)l * DD, ln1_b + (long)l * DD, HA);

        long nqkv = 3L * DD * DD;
        cvt_f32_f16<<<(int)((nqkv + TPB - 1) / TPB), TPB, 0, stream>>>(qkv_w + (long)l * nqkv, WL, nqkv);
        wmma_gemm<<<gemm_blocks(NTOK, 3 * DD), TPB, 0, stream>>>(
            HA, WL, qkv_b + (long)l * 3 * DD, nullptr, QKV, nullptr,
            NTOK, 3 * DD, DD, 3 * DD, 0, 0);

        attn_kernel<<<dim3(TT, HH, NB), TPB, 0, stream>>>(QKV, HA);

        long nproj = (long)DD * DD;
        cvt_f32_f16<<<(int)((nproj + TPB - 1) / TPB), TPB, 0, stream>>>(proj_w + (long)l * nproj, WL, nproj);
        wmma_gemm<<<gemm_blocks(NTOK, DD), TPB, 0, stream>>>(
            HA, WL, proj_b + (long)l * DD, X, X, nullptr,
            NTOK, DD, DD, DD, 0, 0);

        // --- MLP half ---
        ln_kernel<<<NTOK, TPB, 0, stream>>>(X, ln2_w + (long)l * DD, ln2_b + (long)l * DD, HA);

        long nfc = (long)FFN * DD;
        cvt_f32_f16<<<(int)((nfc + TPB - 1) / TPB), TPB, 0, stream>>>(fc1_w + (long)l * nfc, WL, nfc);
        wmma_gemm<<<gemm_blocks(NTOK, FFN), TPB, 0, stream>>>(
            HA, WL, fc1_b + (long)l * FFN, nullptr, nullptr, HB,
            NTOK, FFN, DD, 0, FFN, /*gelu=*/1);

        cvt_f32_f16<<<(int)((nfc + TPB - 1) / TPB), TPB, 0, stream>>>(fc2_w + (long)l * nfc, WL, nfc);
        wmma_gemm<<<gemm_blocks(NTOK, DD), TPB, 0, stream>>>(
            HB, WL, fc2_b + (long)l * DD, X, X, nullptr,
            NTOK, DD, FFN, DD, 0, 0);
    }

    // final LN + tied LM head
    ln_kernel<<<NTOK, TPB, 0, stream>>>(X, lnf_w, lnf_b, HA);
    {
        long nemb = (long)VV * DD;
        cvt_f32_f16<<<(int)((nemb + TPB - 1) / TPB), TPB, 0, stream>>>(tok_emb, WEMB, nemb);
    }
    wmma_gemm<<<gemm_blocks(NTOK, VV), TPB, 0, stream>>>(
        HA, WEMB, nullptr, nullptr, logits, nullptr,
        NTOK, VV, DD, VV, 0, 0);

    // loss
    loss_init<<<1, 1, 0, stream>>>(ACC);
    loss_rows<<<NTOK, TPB, 0, stream>>>(logits, targets, ACC);
    loss_final<<<1, 1, 0, stream>>>(ACC, lossO);
}
